// HelformerAttention_44006234915551
// MI455X (gfx1250) — compile-verified, matrix operand order checked by reading, not run on hardware
//
#include <hip/hip_runtime.h>

// ---------------------------------------------------------------------------
// Types for CDNA5 WMMA (gfx1250, wave32)
// ---------------------------------------------------------------------------
typedef __bf16 bf16_t;
typedef bf16_t v16bf __attribute__((ext_vector_type(16)));
typedef bf16_t v8bf  __attribute__((ext_vector_type(8)));
typedef float  v8f   __attribute__((ext_vector_type(8)));
typedef unsigned int u32x4 __attribute__((ext_vector_type(4)));
typedef int i32x4 __attribute__((ext_vector_type(4)));
typedef int i32x8 __attribute__((ext_vector_type(8)));

union BFrag { v16bf v; v8bf h[2]; };

#if defined(__has_builtin)
#if __has_builtin(__builtin_amdgcn_tensor_load_to_lds) && __has_builtin(__builtin_amdgcn_s_wait_tensorcnt)
#define HAVE_TDM 1
#endif
#endif

// fp32 -> bf16, round-to-nearest-even
__device__ __forceinline__ bf16_t f2bf(float f) {
    unsigned u = __builtin_bit_cast(unsigned, f);
    unsigned r = (u + 0x7FFFu + ((u >> 16) & 1u)) >> 16;
    unsigned short s = (unsigned short)r;
    return __builtin_bit_cast(bf16_t, s);
}

__device__ __forceinline__ float sigm(float v) {
    return 1.0f / (1.0f + __expf(-v));
}

// 16-bit A/B fragment loader for V_WMMA_*_16X16X32 operands.
// ISA layout: lane L holds row (L&15); its 16 values cover K =
// {0..7,16..23} (lanes 0-15) or {8..15,24..31} (lanes 16-31) of the 32-K slab.
// With row-major bf16 storage this is two contiguous 16B loads per lane.
__device__ __forceinline__ v16bf frag_ld(const bf16_t* base, int ld,
                                         int row0, int k0, int lane) {
    int r = row0 + (lane & 15);
    int c = k0 + ((lane & 16) >> 1);     // +8 for upper half-wave
    const bf16_t* p = base + (size_t)r * ld + c;
    BFrag f;
    f.h[0] = *(const v8bf*)(p);
    f.h[1] = *(const v8bf*)(p + 16);
    return f.v;
}

__device__ __forceinline__ v8f wmma_bf16(v16bf a, v16bf b, v8f c) {
    // D(16x16 f32) = A(16x32 bf16) * B(32x16 bf16) + C
    return __builtin_amdgcn_wmma_f32_16x16x32_bf16(
        /*neg_a=*/false, a, /*neg_b=*/false, b,
        /*c_mod=*/(short)0, c, /*reuse_a=*/false, /*reuse_b=*/false);
}

#ifdef HAVE_TDM
// Issue one 2D Tensor-Data-Mover load: tile (tile_d1 rows x tile_d0 elems) of
// 2-byte elements from a row-major tensor (row stride = stride_elems) into LDS
// at byte offset lds_off. Descriptor per CDNA5 ISA.
__device__ __forceinline__ void tdm_load_2d_b16(unsigned lds_off, const void* gptr,
                                                unsigned tensor_d0, unsigned tensor_d1,
                                                unsigned tile_d0, unsigned tile_d1,
                                                unsigned stride_elems) {
    unsigned long long ga = (unsigned long long)gptr;
    u32x4 g0 = {};
    g0[0] = 1u;                                        // count=1, user D#
    g0[1] = lds_off;                                   // LDS byte address
    g0[2] = (unsigned)(ga & 0xffffffffu);              // global_addr[31:0]
    g0[3] = (unsigned)((ga >> 32) & 0x1ffffffu)        // global_addr[56:32]
            | (2u << 30);                              // type = 2 (image)
    i32x8 g1 = {};
    g1[0] = (int)0x00010000u;                          // data_size=1 -> 2 bytes
    g1[1] = (int)((tensor_d0 & 0xffffu) << 16);        // tensor_dim0[15:0]
    g1[2] = (int)(((tensor_d0 >> 16) & 0xffffu) |
                  ((tensor_d1 & 0xffffu) << 16));      // d0 hi | tensor_dim1 lo
    g1[3] = (int)(((tensor_d1 >> 16) & 0xffffu) |
                  ((tile_d0 & 0xffffu) << 16));        // d1 hi | tile_dim0
    g1[4] = (int)(tile_d1 & 0xffffu);                  // tile_dim1, tile_dim2=0
    g1[5] = (int)stride_elems;                         // tensor_dim0_stride lo
    g1[6] = 0;
    g1[7] = 0;
    i32x4 z4 = {};
#if __clang_major__ >= 23
    i32x8 z8 = {};
    __builtin_amdgcn_tensor_load_to_lds(g0, g1, z4, z4, z8, 0);
#else
    __builtin_amdgcn_tensor_load_to_lds(g0, g1, z4, z4, 0);
#endif
}
#endif

// ---------------------------------------------------------------------------
// Problem constants
// ---------------------------------------------------------------------------
#define E_  1024
#define H_  16
#define D_  64
#define NS_ 12
#define B_  8
#define S_  512
#define MTOK (B_ * S_)          // 4096 token rows

// d_out layout (tuple flattened): out, attn, level_f, trend_f, hw_features
#define OFF_OUT  ((size_t)0)
#define OFF_ATTN ((size_t)(B_ * S_ * E_))                     //  4,194,304
#define OFF_LVL  (OFF_ATTN + (size_t)B_ * H_ * S_ * S_)       // 37,748,736
#define OFF_TRD  (OFF_LVL + (size_t)B_ * E_)
#define OFF_HW   (OFF_TRD + (size_t)B_ * E_)

// ---------------------------------------------------------------------------
// Kernel 1: convert 7 fp32 [E,E] weights to bf16 row-major (L2-resident)
// ---------------------------------------------------------------------------
__global__ void convert_w_kernel(const float* w0, const float* w1, const float* w2,
                                 const float* w3, const float* w4, const float* w5,
                                 const float* w6, bf16_t* dst) {
    int mat = blockIdx.y;
    const float* src;
    switch (mat) {
        case 0: src = w0; break;
        case 1: src = w1; break;
        case 2: src = w2; break;
        case 3: src = w3; break;
        case 4: src = w4; break;
        case 5: src = w5; break;
        default: src = w6; break;
    }
    bf16_t* d = dst + (size_t)mat * E_ * E_;
    for (int i = blockIdx.x * blockDim.x + threadIdx.x; i < E_ * E_;
         i += gridDim.x * blockDim.x)
        d[i] = f2bf(src[i]);
}

// ---------------------------------------------------------------------------
// Kernel 2: Holt-Winters scan. ONE workgroup, 32 waves, all recurrent state
// resident in LDS (~304KB of the WGP's 320KB). Per step: 3 GEMMs
// out.T = W(1024x1024) @ act.T(1024x8pad16); each wave owns 2 of 64 M-tiles.
// Next step's x slab is prefetched (global_prefetch_b8) during prep.
// ---------------------------------------------------------------------------
#define SM_LVL  0
#define SM_TRD  (32 * 1024)
#define SM_SEAS (64 * 1024)
#define SM_XT   (112 * 1024)
#define SM_LN   (144 * 1024)
#define SM_TN   (176 * 1024)
#define SM_SU   (208 * 1024)
#define SM_B1   (240 * 1024)    // bf16 [16][1024], rows 8..15 zero pad
#define SM_B2   (272 * 1024)
#define SM_SI   (304 * 1024)
#define SCAN_SMEM (304 * 1024 + 16)

__device__ __forceinline__ void scan_gemm(const bf16_t* __restrict__ W,
                                          const float* __restrict__ bias,
                                          const bf16_t* bin,   // LDS [16][1024]
                                          float* outb,         // LDS [8][1024]
                                          int wv, int lane) {
    v8f acc0 = {};
    v8f acc1 = {};
    int row0 = wv * 32;                  // 2 consecutive m-tiles per wave
    #pragma unroll 4
    for (int kt = 0; kt < 32; ++kt) {
        int k0 = kt * 32;
        v16bf bfr = frag_ld(bin, E_, 0, k0, lane);          // B: act.T (ds)
        v16bf a0  = frag_ld(W, E_, row0, k0, lane);         // A: weight rows
        v16bf a1  = frag_ld(W, E_, row0 + 16, k0, lane);
        acc0 = wmma_bf16(a0, bfr, acc0);
        acc1 = wmma_bf16(a1, bfr, acc1);
    }
    int n  = lane & 15;                  // batch column
    int mo = (lane & 16) >> 1;           // +8 row for upper half-wave
    if (n < 8) {
        #pragma unroll
        for (int v = 0; v < 8; ++v) {
            int m0 = row0 + v + mo;
            outb[n * E_ + m0] = acc0[v] + bias[m0];
            int m1 = row0 + 16 + v + mo;
            outb[n * E_ + m1] = acc1[v] + bias[m1];
        }
    }
}

__global__ void __launch_bounds__(1024, 1)
scan_kernel(const float* __restrict__ x, const float* __restrict__ lvl0,
            const float* __restrict__ trd0, const int* __restrict__ tidx,
            const float* ap, const float* bp, const float* gp,
            const bf16_t* __restrict__ Wl, const float* __restrict__ bl,
            const bf16_t* __restrict__ Wt, const float* __restrict__ bt,
            const bf16_t* __restrict__ Wsw, const float* __restrict__ bs,
            const float* __restrict__ seas0,
            float* __restrict__ hw_out, float* __restrict__ lvl_out,
            float* __restrict__ trd_out) {
    extern __shared__ char smem[];
    float*  lvl  = (float*)(smem + SM_LVL);
    float*  trd  = (float*)(smem + SM_TRD);
    float*  seas = (float*)(smem + SM_SEAS);
    float*  xt   = (float*)(smem + SM_XT);
    float*  ln   = (float*)(smem + SM_LN);
    float*  tn   = (float*)(smem + SM_TN);
    float*  su   = (float*)(smem + SM_SU);
    bf16_t* b1   = (bf16_t*)(smem + SM_B1);
    bf16_t* b2   = (bf16_t*)(smem + SM_B2);
    int*    si_sh = (int*)(smem + SM_SI);

    int tid = threadIdx.x, wv = tid >> 5, lane = tid & 31;
    float a  = sigm(ap[0]);
    float be = sigm(bp[0]);
    float g  = sigm(gp[0]);

    for (int i = tid; i < B_ * E_; i += 1024) { lvl[i] = lvl0[i]; trd[i] = trd0[i]; }
    for (int i = tid; i < NS_ * E_; i += 1024) seas[i] = seas0[i];
    bf16_t z = f2bf(0.0f);
    for (int i = tid; i < B_ * E_; i += 1024) {   // zero the M-pad rows 8..15
        b1[B_ * E_ + i] = z;
        b2[B_ * E_ + i] = z;
    }
    __syncthreads();

    for (int s = 0; s < S_; ++s) {
        if (tid == 0) {
            int sum = 0;
            for (int bb = 0; bb < B_; ++bb) sum += tidx[bb * S_ + s];
            int si = (int)floorf((float)sum / (float)B_);
            *si_sh = ((si % NS_) + NS_) % NS_;
        }
        __syncthreads();
        int si = *si_sh;

        // prep1: u = a*(x - seas[si]) + (1-a)*(level + trend)
        for (int i = tid; i < B_ * E_; i += 1024) {
            int bb = i >> 10, e = i & (E_ - 1);
            float xv = x[((size_t)bb * S_ + s) * E_ + e];
            xt[i] = xv;
            float des = xv - seas[si * E_ + e];
            float u = a * des + (1.0f - a) * (lvl[i] + trd[i]);
            b1[i] = f2bf(u);
            if (s + 1 < S_)   // hide next step's x latency behind the 3 GEMMs
                __builtin_prefetch(&x[((size_t)bb * S_ + (s + 1)) * E_ + e], 0, 3);
        }
        __syncthreads();

        scan_gemm(Wl, bl, b1, ln, wv, lane);        // level_new
        __syncthreads();

        // prep2: trend input + seasonal input
        for (int i = tid; i < B_ * E_; i += 1024) {
            int e = i & (E_ - 1);
            float lnv = ln[i];
            b1[i] = f2bf(be * (lnv - lvl[i]) + (1.0f - be) * trd[i]);
            b2[i] = f2bf(g * (xt[i] - lnv) + (1.0f - g) * seas[si * E_ + e]);
        }
        __syncthreads();

        scan_gemm(Wt,  bt, b1, tn, wv, lane);       // trend_new
        scan_gemm(Wsw, bs, b2, su, wv, lane);       // seasonal_update
        __syncthreads();

        // epilogue: hw out, carry update, seasonal mean update
        for (int i = tid; i < B_ * E_; i += 1024) {
            int bb = i >> 10, e = i & (E_ - 1);
            float lnv = ln[i], tnv = tn[i], suv = su[i];
            hw_out[((size_t)bb * S_ + s) * E_ + e] = lnv + tnv + suv;
            lvl[i] = lnv;
            trd[i] = tnv;
        }
        {
            int e = tid;
            if (e < E_) {
                float sm = 0.0f;
                #pragma unroll
                for (int bb = 0; bb < B_; ++bb) sm += su[bb * E_ + e];
                seas[si * E_ + e] = sm * 0.125f;
            }
        }
        __syncthreads();
    }

    for (int i = tid; i < B_ * E_; i += 1024) {
        lvl_out[i] = lvl[i];
        trd_out[i] = trd[i];
    }
}

// ---------------------------------------------------------------------------
// Kernel 3: enhanced = hw + temporal_emb[tidx % NS], converted to bf16
// ---------------------------------------------------------------------------
__global__ void enhance_kernel(const float* __restrict__ hw,
                               const float* __restrict__ temb,
                               const int* __restrict__ tidx,
                               bf16_t* __restrict__ enh) {
    for (size_t i = (size_t)blockIdx.x * blockDim.x + threadIdx.x;
         i < (size_t)MTOK * E_; i += (size_t)gridDim.x * blockDim.x) {
        int e = (int)(i & (E_ - 1));
        int m = (int)(i >> 10);
        int s = m & (S_ - 1), b = m >> 9;
        int ti = tidx[b * S_ + s];
        int r = ((ti % NS_) + NS_) % NS_;
        enh[i] = f2bf(hw[i] + temb[r * E_ + e]);
    }
}

// ---------------------------------------------------------------------------
// Kernel 4: fused Q/K/V projections. One wave = one 16M x 64N strip for all
// three matrices (A fragment reused 12x per K slab). V written transposed
// per head: Vt[b][h][d][s] for contiguous PV B-fragments.
// ---------------------------------------------------------------------------
__global__ void qkv_kernel(const bf16_t* __restrict__ Enh,
                           const bf16_t* __restrict__ Wq, const bf16_t* __restrict__ Wk,
                           const bf16_t* __restrict__ Wv,
                           const float* __restrict__ bq, const float* __restrict__ bk,
                           const float* __restrict__ bv,
                           bf16_t* __restrict__ Qb, bf16_t* __restrict__ Kb,
                           bf16_t* __restrict__ Vt) {
    int tid = threadIdx.x, wv = tid >> 5, lane = tid & 31;
    int strip = blockIdx.x * 8 + wv;         // 4096 strips
    int mt = strip >> 4;                     // token tile 0..255
    int n0 = (strip & 15) * 64;              // feature base

    v8f zero = {};
    v8f acc[3][4];
    #pragma unroll
    for (int ms = 0; ms < 3; ++ms)
        #pragma unroll
        for (int nt = 0; nt < 4; ++nt) acc[ms][nt] = zero;

    for (int kt = 0; kt < 32; ++kt) {
        int k0 = kt * 32;
        v16bf afr = frag_ld(Enh, E_, mt * 16, k0, lane);
        #pragma unroll
        for (int nt = 0; nt < 4; ++nt) {
            acc[0][nt] = wmma_bf16(afr, frag_ld(Wq, E_, n0 + nt * 16, k0, lane), acc[0][nt]);
            acc[1][nt] = wmma_bf16(afr, frag_ld(Wk, E_, n0 + nt * 16, k0, lane), acc[1][nt]);
            acc[2][nt] = wmma_bf16(afr, frag_ld(Wv, E_, n0 + nt * 16, k0, lane), acc[2][nt]);
        }
    }

    int nl = lane & 15, mo = (lane & 16) >> 1;
    #pragma unroll
    for (int nt = 0; nt < 4; ++nt) {
        #pragma unroll
        for (int v = 0; v < 8; ++v) {
            int m = mt * 16 + v + mo;        // token row
            int n = n0 + nt * 16 + nl;       // feature
            int b = m >> 9, s = m & (S_ - 1);
            Qb[(size_t)m * E_ + n] = f2bf(acc[0][nt][v] + bq[n]);
            Kb[(size_t)m * E_ + n] = f2bf(acc[1][nt][v] + bk[n]);
            int h = n >> 6, d = n & (D_ - 1);
            Vt[(((size_t)b * H_ + h) * D_ + d) * S_ + s] = f2bf(acc[2][nt][v] + bv[n]);
        }
    }
}

// ---------------------------------------------------------------------------
// Kernel 5: attention. One workgroup per (b, h, 64-row q block).
// K head-slab (512x64 bf16, 64KB) staged into LDS by the Tensor Data Mover
// (one 2D descriptor; s_wait_tensorcnt) -> removes the 4x inter-wave re-read
// of K fragments from L2. V slab staged the same way into the scores region
// (free after softmax). scores (WMMA) -> softmax -> attn fp32 out + bf16 LDS
// copy -> PV (WMMA).
// LDS: 128KB scores/V + 64KB attn bf16 + 64KB K + reductions (~258KB).
// ---------------------------------------------------------------------------
#define ATT_SC   0
#define ATT_AB   (64 * 512 * 4)                 // 131072
#define ATT_KST  (ATT_AB + 64 * 512 * 2)        // 196608
#define ATT_REDM (ATT_KST + 512 * 64 * 2)       // 262144
#define ATT_REDS (ATT_REDM + 64 * 4 * 4)
#define ATT_INV  (ATT_REDS + 64 * 4 * 4)
#define ATTN_SMEM (ATT_INV + 64 * 4)

__global__ void attn_kernel(const bf16_t* __restrict__ Qb,
                            const bf16_t* __restrict__ Kb,
                            const bf16_t* __restrict__ Vt,
                            float* __restrict__ attn_out,
                            bf16_t* __restrict__ AOb) {
    extern __shared__ char smem[];
    float*  sc   = (float*)(smem + ATT_SC);      // [64][512] scores, later V
    bf16_t* ab   = (bf16_t*)(smem + ATT_AB);     // [64][512] bf16 attn
    bf16_t* kst  = (bf16_t*)(smem + ATT_KST);    // [512][64] K head slab
    float*  redm = (float*)(smem + ATT_REDM);    // [64][4]
    float*  reds = (float*)(smem + ATT_REDS);    // [64][4]
    float*  inv  = (float*)(smem + ATT_INV);     // [64]

    int blk = blockIdx.x;
    int b = blk >> 7;
    int h = (blk >> 3) & (H_ - 1);
    int q0 = (blk & 7) * 64;
    int tid = threadIdx.x, wv = tid >> 5, lane = tid & 31;

    const bf16_t* Qh = Qb + (size_t)b * S_ * E_ + h * D_;
    const bf16_t* Kh = Kb + (size_t)b * S_ * E_ + h * D_;
    const bf16_t* Vh = Vt + ((size_t)(b * H_ + h) * D_) * S_;   // [d][s]

    // ---- stage K slab into LDS (Tensor Data Mover) ----
#ifdef HAVE_TDM
    if (wv == 0) {
        // tile: 512 rows x 64 elems (2B), row stride 1024 elems
        tdm_load_2d_b16(ATT_KST, Kh, 64, 512, 64, 512, E_);
        __builtin_amdgcn_s_wait_tensorcnt(0);
    }
#else
    for (int c = tid; c < 512 * 4; c += 256) {   // 16B chunks, 4 per row
        int row = c >> 2, coff = (c & 3) * 8;
        *(v8bf*)(kst + row * 64 + coff) = *(const v8bf*)(Kh + (size_t)row * E_ + coff);
    }
#endif
    // Q fragments (reused for all 16 key tiles of this wave)
    int qt = wv & 3, half = wv >> 2;        // wave -> (q tile, 256-col half)
    v16bf aq0 = frag_ld(Qh, E_, q0 + qt * 16, 0, lane);
    v16bf aq1 = frag_ld(Qh, E_, q0 + qt * 16, 32, lane);
    __syncthreads();

    // ---- phase 1: scores = Q K^T * 1/sqrt(D) ----
    for (int nt = 0; nt < 16; ++nt) {
        int krow = half * 256 + nt * 16;
        v8f acc = {};
        acc = wmma_bf16(aq0, frag_ld(kst, D_, krow, 0, lane), acc);
        acc = wmma_bf16(aq1, frag_ld(kst, D_, krow, 32, lane), acc);
        int n = lane & 15, mo = (lane & 16) >> 1;
        #pragma unroll
        for (int v = 0; v < 8; ++v) {
            int qrow = qt * 16 + v + mo;
            sc[qrow * 512 + krow + n] = acc[v] * 0.125f;
        }
    }
    __syncthreads();

    // ---- phase 2: softmax over 512 columns per row ----
    {
        int r = tid & 63, qtr = tid >> 6;
        float* row = sc + r * 512 + qtr * 128;
        float mx = -3.4e38f;
        for (int i = 0; i < 128; ++i) mx = fmaxf(mx, row[i]);
        redm[r * 4 + qtr] = mx;
        __syncthreads();
        float m4 = fmaxf(fmaxf(redm[r * 4], redm[r * 4 + 1]),
                         fmaxf(redm[r * 4 + 2], redm[r * 4 + 3]));
        float sm = 0.0f;
        for (int i = 0; i < 128; ++i) {
            float e = __expf(row[i] - m4);
            row[i] = e;
            sm += e;
        }
        reds[r * 4 + qtr] = sm;
        __syncthreads();
        if (qtr == 0)
            inv[r] = 1.0f / (reds[r * 4] + reds[r * 4 + 1] +
                             reds[r * 4 + 2] + reds[r * 4 + 3]);
    }
    __syncthreads();

    // normalize; write fp32 attn output (coalesced) + bf16 LDS copy for PV
    float* attn_base = attn_out + (((size_t)(b * H_ + h) * S_ + q0) * S_);
    for (int i = tid; i < 64 * 512; i += 256) {
        int r = i >> 9;
        float p = sc[i] * inv[r];
        attn_base[(size_t)r * S_ + (i & 511)] = p;
        ab[i] = f2bf(p);
    }
    __syncthreads();

    // ---- stage V slab into the (now free) scores region ----
    bf16_t* vst = (bf16_t*)(smem + ATT_SC);      // [64][512]
#ifdef HAVE_TDM
    if (wv == 0) {
        // tile: 64 rows x 512 elems (2B), row stride 512 elems (contiguous)
        tdm_load_2d_b16(ATT_SC, Vh, 512, 64, 512, 64, S_);
        __builtin_amdgcn_s_wait_tensorcnt(0);
    }
#else
    for (int c = tid; c < 64 * 32; c += 256) {   // 16B chunks, 32 per row
        int row = c >> 5, coff = (c & 31) * 8;
        *(v8bf*)(vst + row * 512 + coff) = *(const v8bf*)(Vh + (size_t)row * S_ + coff);
    }
#endif
    __syncthreads();

    // ---- phase 3: out = attn @ V ----
    for (int tt = wv; tt < 16; tt += 8) {
        int qt3 = tt & 3, dt = tt >> 2;
        v8f acc = {};
        for (int kt = 0; kt < 16; ++kt) {
            int k0 = kt * 32;
            v16bf af  = frag_ld(ab, 512, qt3 * 16, k0, lane);   // LDS attn
            v16bf bfv = frag_ld(vst, 512, dt * 16, k0, lane);   // LDS V^T
            acc = wmma_bf16(af, bfv, acc);
        }
        int n = lane & 15, mo = (lane & 16) >> 1;
        #pragma unroll
        for (int v = 0; v < 8; ++v) {
            int q = q0 + qt3 * 16 + v + mo;
            int d = dt * 16 + n;
            AOb[((size_t)b * S_ + q) * E_ + h * D_ + d] = f2bf(acc[v]);
        }
    }
}

// ---------------------------------------------------------------------------
// Kernel 6: final projection out = AO @ Wo.T + bo  (fp32 output)
// ---------------------------------------------------------------------------
__global__ void proj_kernel(const bf16_t* __restrict__ A,
                            const bf16_t* __restrict__ Wo,
                            const float* __restrict__ bo,
                            float* __restrict__ out) {
    int tid = threadIdx.x, wv = tid >> 5, lane = tid & 31;
    int strip = blockIdx.x * 8 + wv;
    int mt = strip >> 4;
    int n0 = (strip & 15) * 64;

    v8f zero = {};
    v8f acc[4];
    #pragma unroll
    for (int nt = 0; nt < 4; ++nt) acc[nt] = zero;

    for (int kt = 0; kt < 32; ++kt) {
        int k0 = kt * 32;
        v16bf af = frag_ld(A, E_, mt * 16, k0, lane);
        #pragma unroll
        for (int nt = 0; nt < 4; ++nt)
            acc[nt] = wmma_bf16(af, frag_ld(Wo, E_, n0 + nt * 16, k0, lane), acc[nt]);
    }

    int nl = lane & 15, mo = (lane & 16) >> 1;
    #pragma unroll
    for (int nt = 0; nt < 4; ++nt)
        #pragma unroll
        for (int v = 0; v < 8; ++v) {
            int m = mt * 16 + v + mo;
            int n = n0 + nt * 16 + nl;
            out[(size_t)m * E_ + n] = acc[nt][v] + bo[n];
        }
}

// ---------------------------------------------------------------------------
// Host launcher
// ---------------------------------------------------------------------------
extern "C" void kernel_launch(void* const* d_in, const int* in_sizes, int n_in,
                              void* d_out, int out_size, void* d_ws, size_t ws_size,
                              hipStream_t stream) {
    (void)in_sizes; (void)n_in; (void)out_size; (void)ws_size;

    const float* x    = (const float*)d_in[0];
    const float* lvl0 = (const float*)d_in[1];
    const float* trd0 = (const float*)d_in[2];
    const int*   tidx = (const int*)d_in[3];
    const float* ap   = (const float*)d_in[4];
    const float* bp   = (const float*)d_in[5];
    const float* gp   = (const float*)d_in[6];
    const float* Wl = (const float*)d_in[7];  const float* bl = (const float*)d_in[8];
    const float* Wt = (const float*)d_in[9];  const float* bt = (const float*)d_in[10];
    const float* Ws = (const float*)d_in[11]; const float* bs = (const float*)d_in[12];
    const float* seas0 = (const float*)d_in[13];
    const float* temb  = (const float*)d_in[14];
    const float* Wq = (const float*)d_in[15]; const float* bq = (const float*)d_in[16];
    const float* Wk = (const float*)d_in[17]; const float* bk = (const float*)d_in[18];
    const float* Wv = (const float*)d_in[19]; const float* bv = (const float*)d_in[20];
    const float* Wo = (const float*)d_in[21]; const float* bo = (const float*)d_in[22];

    float* out = (float*)d_out;

    // Workspace layout (bf16 elements): 7 weights + enhanced + Q + K + V^T + AO
    bf16_t* wsb   = (bf16_t*)d_ws;
    bf16_t* Wl_b  = wsb + (size_t)0 * E_ * E_;
    bf16_t* Wt_b  = wsb + (size_t)1 * E_ * E_;
    bf16_t* Ws_b  = wsb + (size_t)2 * E_ * E_;
    bf16_t* Wq_b  = wsb + (size_t)3 * E_ * E_;
    bf16_t* Wk_b  = wsb + (size_t)4 * E_ * E_;
    bf16_t* Wv_b  = wsb + (size_t)5 * E_ * E_;
    bf16_t* Wo_b  = wsb + (size_t)6 * E_ * E_;
    bf16_t* enh   = wsb + (size_t)7 * E_ * E_;
    bf16_t* Qb    = enh + (size_t)MTOK * E_;
    bf16_t* Kb    = Qb  + (size_t)MTOK * E_;
    bf16_t* Vt    = Kb  + (size_t)MTOK * E_;
    bf16_t* AOb   = Vt  + (size_t)MTOK * E_;

    // 1) weights -> bf16 (live in L2 afterwards)
    convert_w_kernel<<<dim3(512, 7), 256, 0, stream>>>(Wl, Wt, Ws, Wq, Wk, Wv, Wo, wsb);

    // 2) sequential Holt-Winters scan, single WGP-resident workgroup
    scan_kernel<<<1, 1024, SCAN_SMEM, stream>>>(
        x, lvl0, trd0, tidx, ap, bp, gp,
        Wl_b, bl, Wt_b, bt, Ws_b, bs, seas0,
        out + OFF_HW, out + OFF_LVL, out + OFF_TRD);

    // 3) enhanced = hw + temporal_emb  (bf16)
    enhance_kernel<<<4096, 256, 0, stream>>>(out + OFF_HW, temb, tidx, enh);

    // 4) fused Q/K/V projections (V stored head-transposed)
    qkv_kernel<<<512, 256, 0, stream>>>(enh, Wq_b, Wk_b, Wv_b, bq, bk, bv, Qb, Kb, Vt);

    // 5) attention: TDM-staged K/V, scores -> softmax -> attn output + PV
    attn_kernel<<<B_ * H_ * (S_ / 64), 256, ATTN_SMEM, stream>>>(
        Qb, Kb, Vt, out + OFF_ATTN, AOb);

    // 6) output projection
    proj_kernel<<<512, 256, 0, stream>>>(AOb, Wo_b, bo, out + OFF_OUT);
}
